// MultiHeadAttentionLayer_46523085750988
// MI455X (gfx1250) — matrix operand order repaired and checked by reading, not compiled, not run on hardware
//
#include <hip/hip_runtime.h>
#include <stdint.h>

// ---------------------------------------------------------------------------
// MHA layer (BERT+RoPE, head-indexed rope quirk preserved) for gfx1250.
// Matmuls via v_wmma_f32_16x16x32_f16; GEMM A-tiles staged with the CDNA5
// async global->LDS path (global_load_async_to_lds_b128 / s_wait_asynccnt).
// ---------------------------------------------------------------------------

typedef _Float16 h16 __attribute__((ext_vector_type(16)));
typedef _Float16 h8  __attribute__((ext_vector_type(8)));
typedef float    f8  __attribute__((ext_vector_type(8)));

#define WMMA_F16(a, b, c) \
  __builtin_amdgcn_wmma_f32_16x16x32_f16(false, (a), false, (b), (short)0, (c), false, false)

static constexpr int BB = 2, SS = 2048, HH = 768, NHH = 12, HD = 64;
static constexpr int MT = BB * SS;            // 4096 token rows

// --- CDNA5 async memory->LDS copy (16 bytes per lane), tracked by ASYNCcnt.
// LDS byte address = low 32 bits of the generic pointer (ISA 10.2 aperture rule).
__device__ __forceinline__ void async_copy_b128(void* lds, const void* gsrc) {
  uint32_t loff = (uint32_t)(uintptr_t)lds;
  uint64_t gaddr = (uint64_t)(uintptr_t)gsrc;
  asm volatile("global_load_async_to_lds_b128 %0, %1, off"
               :: "v"(loff), "v"(gaddr)
               : "memory");
}
__device__ __forceinline__ void async_wait0() {
  asm volatile("s_wait_asynccnt 0x0" ::: "memory");
}

// --- A fragment: 16x32 f16 tile, row-major source with leading dim `ld` (halves)
// lane L: m = L&15, g = L>>4; halves 0..7 = K[g*8 .. g*8+7], halves 8..15 = K[16+g*8 ..]
__device__ __forceinline__ h16 load_a_frag(const _Float16* p0, int ld) {
  int lane = threadIdx.x & 31;
  int m = lane & 15, g = lane >> 4;
  const _Float16* p = p0 + (size_t)m * ld;
  h8 lo = *(const h8*)(p + g * 8);
  h8 hi = *(const h8*)(p + 16 + g * 8);
  h16 a;
#pragma unroll
  for (int i = 0; i < 8; ++i) { a[i] = lo[i]; a[i + 8] = hi[i]; }
  return a;
}

// --- B fragment: 32x16 f16 (K x N), sourced from B-transposed rows Bt[n][k], ld in halves
// lane L: n = L&15, g = L>>4; halves 0..7 = K[g*8 .. g*8+7], halves 8..15 = K[16+g*8 ..]
__device__ __forceinline__ h16 load_b_frag(const _Float16* bt, int ld) {
  int lane = threadIdx.x & 31;
  int n = lane & 15, g = lane >> 4;
  const _Float16* p = bt + (size_t)n * ld;
  h8 lo = *(const h8*)(p + g * 8);
  h8 hi = *(const h8*)(p + 16 + g * 8);
  h16 b;
#pragma unroll
  for (int i = 0; i < 8; ++i) { b[i] = lo[i]; b[i + 8] = hi[i]; }
  return b;
}

// ---------------------------------------------------------------------------
// f32 -> f16 convert
// ---------------------------------------------------------------------------
__global__ __launch_bounds__(256) void cvt_f16_kernel(const float* __restrict__ in,
                                                      _Float16* __restrict__ out, int n) {
  int i = blockIdx.x * 256 + threadIdx.x;
  if (i < n) out[i] = (_Float16)in[i];
}

// ---------------------------------------------------------------------------
// Tiled WMMA GEMM: C[M][N] = A[M][K] @ W[K][N] + bias.  Tile 128x64, 8 waves,
// each wave 32x32 (2x2 of 16x16).  OUT_F32 selects f32 (final) vs f16 output.
// M%128==0, N%64==0, K%32==0.
// ---------------------------------------------------------------------------
template <bool OUT_F32>
__global__ __launch_bounds__(256) void gemm_kernel(const _Float16* __restrict__ A,
                                                   const _Float16* __restrict__ W,
                                                   const float* __restrict__ bias,
                                                   void* __restrict__ Cout,
                                                   int M, int N, int K) {
  __shared__ __attribute__((aligned(32))) _Float16 As[128 * 32];
  __shared__ __attribute__((aligned(32))) _Float16 Bt[64 * 32];

  const int tid = threadIdx.x;
  const int wave = tid >> 5, lane = tid & 31;
  const int m0 = blockIdx.x * 128;
  const int n0 = blockIdx.y * 64;
  const int wm = wave & 3, wn = wave >> 2;   // 4 x 2 wave grid

  f8 acc[2][2] = {};

  for (int k0 = 0; k0 < K; k0 += 32) {
    // A tile 128x32: async global->LDS, 2 x 16B per thread (row = tid/2)
    {
      int r = tid >> 1;
      int c = (tid & 1) * 16;
      const _Float16* gsrc = A + (size_t)(m0 + r) * K + k0 + c;
      _Float16* ldst = As + r * 32 + c;
      async_copy_b128(ldst, gsrc);
      async_copy_b128(ldst + 8, gsrc + 8);
    }
    // B tile 32x64 -> transposed into Bt[n][k]
    {
      int k = tid >> 3;
      int nc = (tid & 7) * 8;
      h8 v = *(const h8*)(W + (size_t)(k0 + k) * N + n0 + nc);
#pragma unroll
      for (int j = 0; j < 8; ++j) Bt[(nc + j) * 32 + k] = v[j];
      if (k0 + 32 < K) __builtin_prefetch(W + (size_t)(k0 + 32 + k) * N + n0 + nc, 0, 1);
    }
    async_wait0();
    __syncthreads();
#pragma unroll
    for (int ms = 0; ms < 2; ++ms) {
      h16 a = load_a_frag(As + (wm * 32 + ms * 16) * 32, 32);
#pragma unroll
      for (int ns = 0; ns < 2; ++ns) {
        h16 b = load_b_frag(Bt + (wn * 32 + ns * 16) * 32, 32);
        acc[ms][ns] = WMMA_F16(a, b, acc[ms][ns]);
      }
    }
    __syncthreads();
  }

  const int cn = lane & 15, g = lane >> 4;
#pragma unroll
  for (int ms = 0; ms < 2; ++ms)
#pragma unroll
    for (int ns = 0; ns < 2; ++ns) {
      int row0 = m0 + wm * 32 + ms * 16 + 8 * g;
      int col = n0 + wn * 32 + ns * 16 + cn;
      float bv = bias[col];
#pragma unroll
      for (int r = 0; r < 8; ++r) {
        float v = acc[ms][ns][r] + bv;
        if (OUT_F32)
          ((float*)Cout)[(size_t)(row0 + r) * N + col] = v;
        else
          ((_Float16*)Cout)[(size_t)(row0 + r) * N + col] = (_Float16)v;
      }
    }
}

// ---------------------------------------------------------------------------
// RoPE (head-indexed, per reference quirk) + head reshape.
// Qr/Kr/Vr: [4096][768] f16.  Out: Qh,Kh [B,NH,S,64]; Vt [B,NH,64,S].
// ---------------------------------------------------------------------------
__global__ __launch_bounds__(256) void rope_reshape_kernel(
    const _Float16* __restrict__ Qr, const _Float16* __restrict__ Kr,
    const _Float16* __restrict__ Vr, const float* __restrict__ cosp,
    const float* __restrict__ sinp, _Float16* __restrict__ Qh,
    _Float16* __restrict__ Kh, _Float16* __restrict__ Vt) {
  int idx = blockIdx.x * 256 + threadIdx.x;
  if (idx >= MT * HH) return;
  int row = idx / HH;              // b*S + s
  int col = idx - row * HH;
  int b = row >> 11, s = row & (SS - 1);
  int n = col >> 6, d = col & 63;  // head, head-dim

  // reference: cos/sin row index = HEAD index (seq_len_dim quirk)
  float c = cosp[n * HD + d];
  float sn = sinp[n * HD + d];
  int pd = (d < 32) ? (2 * d + 1) : (2 * (d - 32));
  float sgn = (d < 32) ? -1.0f : 1.0f;

  size_t base = (size_t)row * HH;
  float q = (float)Qr[base + col];
  float qp = (float)Qr[base + n * HD + pd];
  float k = (float)Kr[base + col];
  float kp = (float)Kr[base + n * HD + pd];

  size_t o = (((size_t)b * NHH + n) * SS + s) * HD + d;
  Qh[o] = (_Float16)(q * c + sgn * qp * sn);
  Kh[o] = (_Float16)(k * c + sgn * kp * sn);
  Vt[(((size_t)b * NHH + n) * HD + d) * SS + s] = Vr[base + col];
}

// ---------------------------------------------------------------------------
// Attention: per block = (b,h, 32 query rows).  Full 32x2048 score block in
// LDS (f16, 128KB of the 320KB WGP LDS), exact 2-pass softmax, then P@V.
// ---------------------------------------------------------------------------
__global__ __launch_bounds__(256) void attn_kernel(const _Float16* __restrict__ Qh,
                                                   const _Float16* __restrict__ Kh,
                                                   const _Float16* __restrict__ Vt,
                                                   const float* __restrict__ mask,
                                                   _Float16* __restrict__ Ctx) {
  __shared__ __attribute__((aligned(32))) _Float16 sc[32 * SS];  // 128 KB
  __shared__ float red[32 * 8];
  __shared__ float rowmax[32];
  __shared__ float rowsum[32];

  const int tid = threadIdx.x;
  const int wave = tid >> 5, lane = tid & 31;
  const int cn = lane & 15, g = lane >> 4;
  const int bh = blockIdx.y;           // b*NH + h
  const int b = bh / NHH, h = bh - b * NHH;
  const int m0 = blockIdx.x * 32;      // query-row tile within S

  const _Float16* Q = Qh + (size_t)bh * SS * HD;
  const _Float16* Kp = Kh + (size_t)bh * SS * HD;
  const _Float16* Vp = Vt + (size_t)bh * HD * SS;
  const float* mrow = mask + (size_t)b * SS;

  // ---- phase 1: scores = Q K^T / 8 + mask  -> sc (f16)
  {
    const int wm = wave & 1, wn = wave >> 1;  // 2(m) x 4(n)
    h16 afrag[2];
#pragma unroll
    for (int ks = 0; ks < 2; ++ks)
      afrag[ks] = load_a_frag(Q + (size_t)(m0 + wm * 16) * HD + ks * 32, HD);

    for (int j0 = 0; j0 < SS; j0 += 64) {
      const int jn = j0 + wn * 16;
      f8 acc = {};
#pragma unroll
      for (int ks = 0; ks < 2; ++ks) {
        h16 bfrag = load_b_frag(Kp + (size_t)jn * HD + ks * 32, HD);  // K rows == B^T
        acc = WMMA_F16(afrag[ks], bfrag, acc);
      }
      const int col = jn + cn;
      const float madd = (1.0f - mrow[col]) * -10000.0f;
      const int rl0 = wm * 16 + 8 * g;
#pragma unroll
      for (int r = 0; r < 8; ++r)
        sc[(size_t)(rl0 + r) * SS + col] = (_Float16)(acc[r] * 0.125f + madd);
    }
  }
  __syncthreads();

  // ---- phase 2: exact softmax over each of the 32 rows (8 threads / row)
  {
    const int row = tid >> 3, ch = tid & 7;
    _Float16* srow = sc + (size_t)row * SS + ch * 256;
    const h8* sv = (const h8*)srow;
    float mx = -1e30f;
    for (int i = 0; i < 32; ++i) {
      h8 v = sv[i];
#pragma unroll
      for (int j = 0; j < 8; ++j) mx = fmaxf(mx, (float)v[j]);
    }
    red[row * 8 + ch] = mx;
    __syncthreads();
    if (ch == 0) {
      float m2 = red[row * 8];
#pragma unroll
      for (int i = 1; i < 8; ++i) m2 = fmaxf(m2, red[row * 8 + i]);
      rowmax[row] = m2;
    }
    __syncthreads();
    const float rm = rowmax[row];
    float sum = 0.0f;
    for (int i = 0; i < 32; ++i) {
      h8 v = sv[i];
#pragma unroll
      for (int j = 0; j < 8; ++j) sum += __expf((float)v[j] - rm);
    }
    red[row * 8 + ch] = sum;
    __syncthreads();
    if (ch == 0) {
      float s2 = 0.0f;
#pragma unroll
      for (int i = 0; i < 8; ++i) s2 += red[row * 8 + i];
      rowsum[row] = s2;
    }
    __syncthreads();
    const float inv = 1.0f / rowsum[row];
    h8* svw = (h8*)srow;
    for (int i = 0; i < 32; ++i) {
      h8 v = svw[i];
#pragma unroll
      for (int j = 0; j < 8; ++j) v[j] = (_Float16)(__expf((float)v[j] - rm) * inv);
      svw[i] = v;
    }
  }
  __syncthreads();

  // ---- phase 3: ctx = P @ V   (P f16 in LDS; Vt rows are B^T, contiguous)
  {
    const int wm = wave & 1, wn = wave >> 1;  // 2(m) x 4(hd-tile)
    f8 acc = {};
    for (int ks = 0; ks < 64; ++ks) {
      h16 a = load_a_frag(sc + (size_t)(wm * 16) * SS + ks * 32, SS);
      h16 bfr = load_b_frag(Vp + (size_t)(wn * 16) * SS + ks * 32, SS);
      acc = WMMA_F16(a, bfr, acc);
    }
    const int rl0 = wm * 16 + 8 * g;
    const size_t colg = (size_t)h * HD + wn * 16 + cn;
#pragma unroll
    for (int r = 0; r < 8; ++r) {
      size_t rowg = (size_t)b * SS + m0 + rl0 + r;
      Ctx[rowg * HH + colg] = (_Float16)acc[r];
    }
  }
}

// ---------------------------------------------------------------------------
// launch
// ---------------------------------------------------------------------------
extern "C" void kernel_launch(void* const* d_in, const int* in_sizes, int n_in,
                              void* d_out, int out_size, void* d_ws, size_t ws_size,
                              hipStream_t stream) {
  (void)in_sizes; (void)n_in; (void)out_size; (void)ws_size;
  const float* hidden = (const float*)d_in[0];
  const float* mask   = (const float*)d_in[1];
  const float* Wq = (const float*)d_in[2];
  const float* bq = (const float*)d_in[3];
  const float* Wk = (const float*)d_in[4];
  const float* bk = (const float*)d_in[5];
  const float* Wv = (const float*)d_in[6];
  const float* bv = (const float*)d_in[7];
  const float* Wo = (const float*)d_in[8];
  const float* bo = (const float*)d_in[9];
  const float* cosp = (const float*)d_in[10];
  const float* sinp = (const float*)d_in[11];
  float* out = (float*)d_out;

  constexpr size_t MSZ = (size_t)MT * HH;   // 3,145,728 elems
  constexpr size_t WSZ = (size_t)HH * HH;   //   589,824 elems
  _Float16* Xh  = (_Float16*)d_ws;
  _Float16* Wqh = Xh + MSZ;
  _Float16* Wkh = Wqh + WSZ;
  _Float16* Wvh = Wkh + WSZ;
  _Float16* Woh = Wvh + WSZ;
  _Float16* Qr  = Woh + WSZ;
  _Float16* Kr  = Qr + MSZ;
  _Float16* Vr  = Kr + MSZ;
  _Float16* Qhd = Vr + MSZ;
  _Float16* Khd = Qhd + MSZ;
  _Float16* Vtd = Khd + MSZ;
  _Float16* Ctx = Vtd + MSZ;

  // 1) f32 -> f16 conversions
  cvt_f16_kernel<<<(int)((MSZ + 255) / 256), 256, 0, stream>>>(hidden, Xh, (int)MSZ);
  cvt_f16_kernel<<<(int)((WSZ + 255) / 256), 256, 0, stream>>>(Wq, Wqh, (int)WSZ);
  cvt_f16_kernel<<<(int)((WSZ + 255) / 256), 256, 0, stream>>>(Wk, Wkh, (int)WSZ);
  cvt_f16_kernel<<<(int)((WSZ + 255) / 256), 256, 0, stream>>>(Wv, Wvh, (int)WSZ);
  cvt_f16_kernel<<<(int)((WSZ + 255) / 256), 256, 0, stream>>>(Wo, Woh, (int)WSZ);

  // 2) Q/K/V projections (WMMA + async LDS staging)
  dim3 ggrid(MT / 128, HH / 64);
  gemm_kernel<false><<<ggrid, 256, 0, stream>>>(Xh, Wqh, bq, Qr, MT, HH, HH);
  gemm_kernel<false><<<ggrid, 256, 0, stream>>>(Xh, Wkh, bk, Kr, MT, HH, HH);
  gemm_kernel<false><<<ggrid, 256, 0, stream>>>(Xh, Wvh, bv, Vr, MT, HH, HH);

  // 3) RoPE (head-indexed) + per-head layout (V transposed)
  rope_reshape_kernel<<<(int)((MSZ + 255) / 256), 256, 0, stream>>>(
      Qr, Kr, Vr, cosp, sinp, Qhd, Khd, Vtd);

  // 4) attention (WMMA scores + LDS softmax + WMMA context)
  attn_kernel<<<dim3(SS / 32, BB * NHH), 256, 0, stream>>>(Qhd, Khd, Vtd, mask, Ctx);

  // 5) output projection (WMMA, f32 out)
  gemm_kernel<true><<<ggrid, 256, 0, stream>>>(Ctx, Woh, bo, out, MT, HH, HH);
}
